// SparseBlock_21801253994847
// MI455X (gfx1250) — matrix operand order repaired and checked by reading, not compiled, not run on hardware
//
#include <hip/hip_runtime.h>
#include <hip/hip_bf16.h>

#define BB 100
#define CC 64
#define HH 64
#define NBK 2240
#define NPOS 64            // positions per 8x8 block
#define COUNT_PER_CH (NBK * NPOS)   // 143360
#define PITCH_B 72         // LDS half pitch for activations (conflict-free)
#define PITCH_H 68         // LDS float pitch for h staging

typedef __attribute__((ext_vector_type(16))) _Float16 v16h;
typedef __attribute__((ext_vector_type(8)))  _Float16 v8h;
typedef __attribute__((ext_vector_type(8)))  float    v8f;
typedef __attribute__((ext_vector_type(4)))  float    v4f;   // native vec for NT ops

__device__ inline v8f zero8() {
    v8f z;
#pragma unroll
    for (int i = 0; i < 8; ++i) z[i] = 0.0f;
    return z;
}

// A-fragment: two contiguous 8-half chunks (K groups of 8, split by lane half)
__device__ inline v16h load_frag2(const _Float16* p0, const _Float16* p1) {
    v8h lo = *(const v8h*)p0;
    v8h hi = *(const v8h*)p1;
    v16h r;
#pragma unroll
    for (int i = 0; i < 8; ++i) { r[i] = lo[i]; r[i + 8] = hi[i]; }
    return r;
}

// B-fragment: 16 contiguous halves (16 K values for this lane half)
__device__ inline v16h load_frag16(const _Float16* p) {
    v8h lo = *(const v8h*)p;
    v8h hi = *(const v8h*)(p + 8);
    v16h r;
#pragma unroll
    for (int i = 0; i < 8; ++i) { r[i] = lo[i]; r[i + 8] = hi[i]; }
    return r;
}

__device__ inline v8f wmma16(v16h a, v16h b, v8f c) {
    return __builtin_amdgcn_wmma_f32_16x16x32_f16(false, a, false, b, (short)0, c,
                                                  false, false);
}

// ---------------------------------------------------------------------------
// Kernel: convert W1 (64x64) and W2 (64,64,3,3) to f16; W2 -> [k9][o][c]
// ---------------------------------------------------------------------------
__global__ __launch_bounds__(256) void k_prep_w(const float* __restrict__ W1,
                                                const float* __restrict__ W2,
                                                _Float16* __restrict__ w1h,
                                                _Float16* __restrict__ w2h) {
    int i = blockIdx.x * 256 + threadIdx.x;
    if (i < 4096) w1h[i] = (_Float16)W1[i];
    int j = i - 4096;
    if (j >= 0 && j < 9 * 4096) {
        int k9 = j >> 12;
        int oc = j & 4095;
        w2h[j] = (_Float16)W2[oc * 9 + k9];   // W2[o][c][ky][kx] -> [k9][o][c]
    }
}

// ---------------------------------------------------------------------------
// Kernel: copy x -> out, 16B, non-temporal both ways (don't evict hbuf/L2)
// ---------------------------------------------------------------------------
__global__ __launch_bounds__(256) void k_copy(const float* __restrict__ x,
                                              float* __restrict__ out) {
    size_t i = (size_t)blockIdx.x * 256 + threadIdx.x;   // in v4f units
    const v4f* src = (const v4f*)x;
    v4f* dst = (v4f*)out;
    v4f v = __builtin_nontemporal_load(&src[i]);
    __builtin_nontemporal_store(v, &dst[i]);
}

// ---------------------------------------------------------------------------
// Kernel: stage1 -- gather block, GEMM1 (1x1 conv) via WMMA, +b1,
//         per-block channel sum/sumsq, write h to hbuf
// ---------------------------------------------------------------------------
__global__ __launch_bounds__(256) void k_stage1(const float* __restrict__ x,
                                                const int* __restrict__ idx,
                                                const _Float16* __restrict__ w1h,
                                                const float* __restrict__ bias1,
                                                float* __restrict__ hbuf,
                                                float* __restrict__ pbs,
                                                float* __restrict__ pbq) {
    __shared__ __align__(16) _Float16 actT[NPOS * PITCH_B];  // [pos][c]
    __shared__ __align__(16) float    hsh[CC * PITCH_H];     // [c][pos]

    const int blk = blockIdx.x;
    const int t = threadIdx.x;
    const int bi = idx[blk * 3 + 0];
    const int ri = idx[blk * 3 + 1];
    const int ci = idx[blk * 3 + 2];
    const float* xbase = x + (size_t)bi * (CC * HH * HH) + (size_t)ri * 8 * HH + ci * 8;

    // gather: 512 rows of 8 floats; 2 rows per thread; store f16 position-major
#pragma unroll
    for (int r = 0; r < 2; ++r) {
        int row = t * 2 + r;           // 0..511
        int c = row >> 3;
        int y = row & 7;
        const float* src = xbase + (size_t)c * (HH * HH) + y * HH;
        v4f f0 = __builtin_nontemporal_load((const v4f*)(src));
        v4f f1 = __builtin_nontemporal_load((const v4f*)(src + 4));
        int p = y * 8;
        actT[(p + 0) * PITCH_B + c] = (_Float16)f0.x;
        actT[(p + 1) * PITCH_B + c] = (_Float16)f0.y;
        actT[(p + 2) * PITCH_B + c] = (_Float16)f0.z;
        actT[(p + 3) * PITCH_B + c] = (_Float16)f0.w;
        actT[(p + 4) * PITCH_B + c] = (_Float16)f1.x;
        actT[(p + 5) * PITCH_B + c] = (_Float16)f1.y;
        actT[(p + 6) * PITCH_B + c] = (_Float16)f1.z;
        actT[(p + 7) * PITCH_B + c] = (_Float16)f1.w;
    }
    __syncthreads();

    const int wave = t >> 5, lane = t & 31;
    const int mt = wave >> 1;          // 0..3 (M tile)
    const int lhi = lane >> 4;         // lane half
    const int n0 = lane & 15;
    const int m = mt * 16 + n0;        // A row for this lane

    // A fragments (W1), K = 0..31 and 32..63
    v16h a0 = load_frag2(w1h + m * 64 + lhi * 16, w1h + m * 64 + lhi * 16 + 8);
    v16h a1 = load_frag2(w1h + m * 64 + 32 + lhi * 16, w1h + m * 64 + 32 + lhi * 16 + 8);

#pragma unroll
    for (int tt = 0; tt < 2; ++tt) {
        int nt = (wave & 1) * 2 + tt;
        int pos = nt * 16 + n0;
        v16h b0 = load_frag16(&actT[pos * PITCH_B + lhi * 16]);
        v16h b1 = load_frag16(&actT[pos * PITCH_B + 32 + lhi * 16]);
        v8f acc = zero8();
        acc = wmma16(a0, b0, acc);
        acc = wmma16(a1, b1, acc);
        int mrow = mt * 16 + lhi * 8;  // C/D layout: lanes 16-31 hold M+8
#pragma unroll
        for (int j = 0; j < 8; ++j)
            hsh[(mrow + j) * PITCH_H + pos] = acc[j] + bias1[mrow + j];
    }
    __syncthreads();

    // write h to hbuf (16B) and per-block channel stats
    {
        int c = t >> 2, q = t & 3;
        v4f* dst = (v4f*)(hbuf + (size_t)blk * 4096 + c * 64 + q * 16);
        const v4f* src = (const v4f*)&hsh[c * PITCH_H + q * 16];
#pragma unroll
        for (int v = 0; v < 4; ++v) dst[v] = src[v];
    }
    if (t < CC) {
        float s = 0.f, q = 0.f;
#pragma unroll 8
        for (int p = 0; p < NPOS; ++p) {
            float v = hsh[t * PITCH_H + p];
            s += v; q += v * v;
        }
        pbs[t * NBK + blk] = s;
        pbq[t * NBK + blk] = q;
    }
}

// ---------------------------------------------------------------------------
// Kernel: deterministic BN reduce + finalize (64 blocks, one per channel)
// ---------------------------------------------------------------------------
__global__ __launch_bounds__(256) void k_bn_reduce(const float* __restrict__ pbs,
                                                   const float* __restrict__ pbq,
                                                   const float* __restrict__ gamma,
                                                   const float* __restrict__ beta,
                                                   float* __restrict__ scale,
                                                   float* __restrict__ shift) {
    __shared__ float ss[256];
    __shared__ float sq[256];
    int c = blockIdx.x, t = threadIdx.x;
    float s = 0.f, q = 0.f;
    for (int i = t; i < NBK; i += 256) {
        s += pbs[c * NBK + i];
        q += pbq[c * NBK + i];
    }
    ss[t] = s; sq[t] = q;
    __syncthreads();
    for (int off = 128; off > 0; off >>= 1) {
        if (t < off) { ss[t] += ss[t + off]; sq[t] += sq[t + off]; }
        __syncthreads();
    }
    if (t == 0) {
        const float inv_n = 1.0f / (float)COUNT_PER_CH;
        float mean = ss[0] * inv_n;
        float var = sq[0] * inv_n - mean * mean;
        float is = rsqrtf(var + 1e-5f);
        float sc = gamma[c] * is;
        scale[c] = sc;
        shift[c] = beta[c] - mean * sc;
    }
}

// ---------------------------------------------------------------------------
// Kernel: stage2 -- BN1+ReLU into padded 10x10 LDS tile, 3x3 conv as 9
//         shifted WMMA GEMMs, +b2, per-block stats, h2 written in place
// ---------------------------------------------------------------------------
__global__ __launch_bounds__(256) void k_stage2(float* __restrict__ hbuf,
                                                const _Float16* __restrict__ w2h,
                                                const float* __restrict__ bias2,
                                                const float* __restrict__ scale1,
                                                const float* __restrict__ shift1,
                                                float* __restrict__ pbs,
                                                float* __restrict__ pbq) {
    __shared__ __align__(16) _Float16 actP[100 * PITCH_B];   // padded [10x10][c]
    __shared__ __align__(16) float    hsh[CC * PITCH_H];

    const int blk = blockIdx.x;
    const int t = threadIdx.x;

    // zero entire padded tile (as dwords)
    {
        unsigned int* z = (unsigned int*)actP;
        for (int i = t; i < 100 * PITCH_B / 2; i += 256) z[i] = 0u;
    }
    __syncthreads();

    // fill interior with relu(h1*scale1 + shift1) as f16
    {
        int c = t >> 2, q = t & 3;
        float sc = scale1[c], sh = shift1[c];
        const v4f* src = (const v4f*)(hbuf + (size_t)blk * 4096 + c * 64 + q * 16);
#pragma unroll
        for (int v = 0; v < 4; ++v) {
            v4f f = src[v];
            int p = q * 16 + v * 4;
#pragma unroll
            for (int e = 0; e < 4; ++e) {
                float val = f[e];
                val = fmaxf(val * sc + sh, 0.0f);
                int pp = p + e;
                int y = pp >> 3, xx = pp & 7;
                actP[((y + 1) * 10 + (xx + 1)) * PITCH_B + c] = (_Float16)val;
            }
        }
    }
    __syncthreads();

    const int wave = t >> 5, lane = t & 31;
    const int mt = wave >> 1;
    const int lhi = lane >> 4;
    const int n0 = lane & 15;
    const int m = mt * 16 + n0;

    const int nt0 = (wave & 1) * 2;
    const int p0 = nt0 * 16 + n0, p1 = (nt0 + 1) * 16 + n0;
    const int y0 = p0 >> 3, x0 = p0 & 7;
    const int y1 = p1 >> 3, x1 = p1 & 7;

    v8f acc0 = zero8();
    v8f acc1 = zero8();
#pragma unroll
    for (int k9 = 0; k9 < 9; ++k9) {
        const int dy = k9 / 3, dx = k9 % 3;
        const _Float16* wb = w2h + k9 * 4096 + m * 64;
        v16h a0 = load_frag2(wb + lhi * 16, wb + lhi * 16 + 8);
        v16h a1 = load_frag2(wb + 32 + lhi * 16, wb + 32 + lhi * 16 + 8);
        int pp0 = (y0 + dy) * 10 + (x0 + dx);
        int pp1 = (y1 + dy) * 10 + (x1 + dx);
        v16h b00 = load_frag16(&actP[pp0 * PITCH_B + lhi * 16]);
        v16h b01 = load_frag16(&actP[pp0 * PITCH_B + 32 + lhi * 16]);
        acc0 = wmma16(a0, b00, acc0);
        acc0 = wmma16(a1, b01, acc0);
        v16h b10 = load_frag16(&actP[pp1 * PITCH_B + lhi * 16]);
        v16h b11 = load_frag16(&actP[pp1 * PITCH_B + 32 + lhi * 16]);
        acc1 = wmma16(a0, b10, acc1);
        acc1 = wmma16(a1, b11, acc1);
    }

    {
        int mrow = mt * 16 + lhi * 8;
#pragma unroll
        for (int j = 0; j < 8; ++j) {
            hsh[(mrow + j) * PITCH_H + p0] = acc0[j] + bias2[mrow + j];
            hsh[(mrow + j) * PITCH_H + p1] = acc1[j] + bias2[mrow + j];
        }
    }
    __syncthreads();

    {
        int c = t >> 2, q = t & 3;
        v4f* dst = (v4f*)(hbuf + (size_t)blk * 4096 + c * 64 + q * 16);
        const v4f* src = (const v4f*)&hsh[c * PITCH_H + q * 16];
#pragma unroll
        for (int v = 0; v < 4; ++v) dst[v] = src[v];
    }
    if (t < CC) {
        float s = 0.f, q = 0.f;
#pragma unroll 8
        for (int p = 0; p < NPOS; ++p) {
            float v = hsh[t * PITCH_H + p];
            s += v; q += v * v;
        }
        pbs[t * NBK + blk] = s;
        pbq[t * NBK + blk] = q;
    }
}

// ---------------------------------------------------------------------------
// Kernel: scatter -- relu(h2*scale2+shift2) back into out at block positions
// ---------------------------------------------------------------------------
__global__ __launch_bounds__(256) void k_scatter(const float* __restrict__ hbuf,
                                                 const int* __restrict__ idx,
                                                 const float* __restrict__ scale2,
                                                 const float* __restrict__ shift2,
                                                 float* __restrict__ out) {
    const int blk = blockIdx.x;
    const int t = threadIdx.x;
    const int bi = idx[blk * 3 + 0];
    const int ri = idx[blk * 3 + 1];
    const int ci = idx[blk * 3 + 2];

    int c = t >> 2, q = t & 3;
    float sc = scale2[c], sh = shift2[c];
    const float* src = hbuf + (size_t)blk * 4096 + c * 64 + q * 16;
    float* obase = out + (size_t)bi * (CC * HH * HH) + (size_t)c * (HH * HH) +
                   (size_t)ri * 8 * HH + ci * 8;
#pragma unroll
    for (int rr = 0; rr < 2; ++rr) {
        int y = q * 2 + rr;
        v4f f0 = *(const v4f*)(src + rr * 8);
        v4f f1 = *(const v4f*)(src + rr * 8 + 4);
        v4f r0, r1;
#pragma unroll
        for (int e = 0; e < 4; ++e) {
            r0[e] = fmaxf(f0[e] * sc + sh, 0.f);
            r1[e] = fmaxf(f1[e] * sc + sh, 0.f);
        }
        __builtin_nontemporal_store(r0, (v4f*)(obase + y * HH));
        __builtin_nontemporal_store(r1, (v4f*)(obase + y * HH + 4));
    }
}

// ---------------------------------------------------------------------------
extern "C" void kernel_launch(void* const* d_in, const int* in_sizes, int n_in,
                              void* d_out, int out_size, void* d_ws, size_t ws_size,
                              hipStream_t stream) {
    const float* x      = (const float*)d_in[0];
    const int*   idx    = (const int*)d_in[1];
    const float* W1     = (const float*)d_in[2];
    const float* b1     = (const float*)d_in[3];
    const float* gamma1 = (const float*)d_in[4];
    const float* beta1  = (const float*)d_in[5];
    const float* W2     = (const float*)d_in[6];
    const float* b2     = (const float*)d_in[7];
    const float* gamma2 = (const float*)d_in[8];
    const float* beta2  = (const float*)d_in[9];
    float* out = (float*)d_out;

    // workspace layout (floats)
    float* wsf = (float*)d_ws;
    float* scale1 = wsf + 0;
    float* shift1 = wsf + 64;
    float* scale2 = wsf + 128;
    float* shift2 = wsf + 192;
    _Float16* w1h = (_Float16*)(wsf + 512);     // 4096 halves
    _Float16* w2h = (_Float16*)(wsf + 2560);    // 36864 halves
    float* pbs  = wsf + 21504;                  // 64*2240
    float* pbq  = wsf + 21504 + 64 * NBK;
    float* hbuf = wsf + 21504 + 2 * 64 * NBK;   // NBK*4096 floats (reused h1->h2)

    k_prep_w<<<160, 256, 0, stream>>>(W1, W2, w1h, w2h);
    k_copy<<<(BB * CC * HH * HH / 4) / 256, 256, 0, stream>>>(x, out);
    k_stage1<<<NBK, 256, 0, stream>>>(x, idx, w1h, b1, hbuf, pbs, pbq);
    k_bn_reduce<<<64, 256, 0, stream>>>(pbs, pbq, gamma1, beta1, scale1, shift1);
    k_stage2<<<NBK, 256, 0, stream>>>(hbuf, w2h, b2, scale1, shift1, pbs, pbq);
    k_bn_reduce<<<64, 256, 0, stream>>>(pbs, pbq, gamma2, beta2, scale2, shift2);
    k_scatter<<<NBK, 256, 0, stream>>>(hbuf, idx, scale2, shift2, out);
}
